// BiLSTM_CRF_34531537060393
// MI455X (gfx1250) — compile-verified, matrix operand order checked by reading, not run on hardware
//
#include <hip/hip_runtime.h>
#include <stdint.h>

// Problem constants (from reference)
#define BB   64
#define TT   1024
#define EE   128
#define HH   256
#define HDIM 128
#define KK   20
#define G4   512          // 4*HD gate width
#define NROW (BB*TT)      // 65536 rows

#define LOG2E 1.4426950408889634f
#define LN2   0.6931471805599453f

typedef __attribute__((ext_vector_type(16))) _Float16 v16h;
typedef __attribute__((ext_vector_type(8)))  _Float16 v8h;
typedef __attribute__((ext_vector_type(8)))  float    v8f;

union V16HU { v16h v; v8h h[2]; };

// Load a 16x32 f16 WMMA fragment (A-layout; B from row-major W[N][K] is identical)
// base points at element (row0, k0); ld in elements.
__device__ __forceinline__ v16h load_frag_f16(const _Float16* base, int ld, int lane) {
  int row  = lane & 15;
  int koff = (lane & 16) ? 8 : 0;
  const _Float16* p = base + (size_t)row * ld + koff;
  V16HU u;
  u.h[0] = *(const v8h*)(p);
  u.h[1] = *(const v8h*)(p + 16);
  return u.v;
}

// Branch-free hardware transcendentals (v_exp_f32 / v_rcp_f32 / v_log_f32).
// These are TRANS ops and co-execute with XDL WMMA on CDNA5.
__device__ __forceinline__ float fexp2(float x) { return __builtin_amdgcn_exp2f(x); }
__device__ __forceinline__ float frcp(float x)  { return __builtin_amdgcn_rcpf(x); }
__device__ __forceinline__ float fexp(float x)  { return fexp2(x * LOG2E); }
__device__ __forceinline__ float flog(float x)  { return __builtin_amdgcn_logf(x) * LN2; }
__device__ __forceinline__ float sigf(float x)  { return frcp(1.0f + fexp2(-LOG2E * x)); }
__device__ __forceinline__ float tanh_fast(float x) {
  // tanh(x) = 1 - 2/(exp(2x)+1); saturates correctly via exp2->inf/0.
  return 1.0f - 2.0f * frcp(1.0f + fexp2((2.0f * LOG2E) * x));
}

// ---------------------------------------------------------------- cast f32->f16
__global__ void cast16_kernel(const float* __restrict__ in, _Float16* __restrict__ out, int n) {
  int i = blockIdx.x * 256 + threadIdx.x;
  if (i < n) out[i] = (_Float16)in[i];
}

// ---------------------------------------------------------------- embedding gather
__global__ void embed_kernel(const int* __restrict__ x, const float* __restrict__ emb,
                             _Float16* __restrict__ out) {
  int row = blockIdx.x;          // b*T + t
  int e   = threadIdx.x;         // 0..127
  int xi  = x[row];
  out[(size_t)row * EE + e] = (_Float16)emb[(size_t)xi * EE + e];
}

// ---------------------------------------------------------------- generic WMMA GEMM
// out[r][n] = act(bias_a[n] (+ bias_b[n]) + sum_k X[r][k]*W[n][k])
// NTPW: 16-col tiles per wave (8 waves cover out_dim).
// TPERM: write n-major per timestep: out[((t*out_dim)+n)*BB + b] (for LSTM pre tiles).
template<int NTPW, int TPERM, int RELU>
__global__ __launch_bounds__(256) void wmma_gemm_kernel(
    const _Float16* __restrict__ X, const _Float16* __restrict__ W,
    const float* __restrict__ bias_a, const float* __restrict__ bias_b,
    _Float16* __restrict__ out, int in_dim, int out_dim)
{
  int lane = threadIdx.x & 31;
  int wid  = threadIdx.x >> 5;
  int mt   = blockIdx.x;                    // M-tile (16 rows of X)
  int n0   = wid * (NTPW << 4);
  int nl   = lane & 15;
  int mh   = (lane & 16) ? 8 : 0;

  v8f acc[NTPW];
  #pragma unroll
  for (int g = 0; g < NTPW; ++g) {
    int n = n0 + (g << 4) + nl;
    float bv = bias_a[n] + (bias_b ? bias_b[n] : 0.0f);
    #pragma unroll
    for (int i = 0; i < 8; ++i) acc[g][i] = bv;
  }

  const _Float16* xrow = X + (size_t)(mt * 16) * in_dim;
  for (int k = 0; k < in_dim; k += 32) {
    v16h a = load_frag_f16(xrow + k, in_dim, lane);
    #pragma unroll
    for (int g = 0; g < NTPW; ++g) {
      v16h b = load_frag_f16(W + (size_t)(n0 + (g << 4)) * in_dim + k, in_dim, lane);
      acc[g] = __builtin_amdgcn_wmma_f32_16x16x32_f16(false, a, false, b, (short)0,
                                                      acc[g], false, false);
    }
  }

  #pragma unroll
  for (int g = 0; g < NTPW; ++g) {
    int n = n0 + (g << 4) + nl;
    #pragma unroll
    for (int i = 0; i < 8; ++i) {
      int r = mt * 16 + mh + i;
      float v = acc[g][i];
      if (RELU) v = fmaxf(v, 0.0f);
      size_t idx;
      if (TPERM) { int b = r >> 10; int t = r & (TT - 1);
                   idx = ((size_t)t * out_dim + n) * BB + b; }     // n-major
      else       { idx = (size_t)r * out_dim + n; }
      out[idx] = (_Float16)v;
    }
  }
}

// ---------------------------------------------------------------- persistent LSTM recurrence
// grid = 8 blocks: dir = blockIdx&1, batch-group = blockIdx>>1 (16 rows each).
// g(t) = pre[t] + h(t-1) @ w_hh^T ; w_hh fragments hoisted to VGPRs; h double-buffered
// in LDS; pre tiles (n-major [512][16] f16 = 16 KB) triple-buffered in LDS via async
// global->LDS copies (ASYNCcnt). One workgroup barrier per timestep.
__global__ __launch_bounds__(256) void lstm_rec_kernel(
    const _Float16* __restrict__ pre_f, const _Float16* __restrict__ pre_r,
    const float* __restrict__ whh_f, const float* __restrict__ whh_r,
    _Float16* __restrict__ out /* [b][t][HH] */)
{
  int dir = blockIdx.x & 1;
  int bg  = blockIdx.x >> 1;
  const _Float16* pre = dir ? pre_r : pre_f;
  const float*    whh = dir ? whh_r : whh_f;
  int coloff = dir ? HDIM : 0;
  int bbase  = bg * 16;

  int tid  = threadIdx.x;
  int lane = tid & 31;
  int wid  = tid >> 5;
  int nl   = lane & 15;
  int hc   = wid * 16 + nl;                 // hidden column 0..127 per wave
  int mh   = (lane & 16) ? 8 : 0;
  int koff = (lane & 16) ? 8 : 0;

  // Hoist all w_hh B-fragments (4 gates x 4 k-tiles) into registers, f32 -> f16.
  v16h Bf[4][4];
  #pragma unroll
  for (int g = 0; g < 4; ++g)
    #pragma unroll
    for (int kt = 0; kt < 4; ++kt) {
      const float* p = whh + (size_t)(g * HDIM + hc) * HDIM + kt * 32 + koff;
      v16h f;
      #pragma unroll
      for (int j = 0; j < 8; ++j) { f[j] = (_Float16)p[j]; f[8 + j] = (_Float16)p[16 + j]; }
      Bf[g][kt] = f;
    }

  __shared__ _Float16 h_lds[2][16 * HDIM];     // double-buffered h (2 x 4 KB)
  __shared__ _Float16 stage[3][G4 * 16];       // triple-buffered n-major pre tiles (3 x 16 KB)
  for (int i = tid; i < 16 * HDIM; i += 256) h_lds[0][i] = (_Float16)0.0f;
  v8f c;
  #pragma unroll
  for (int i = 0; i < 8; ++i) c[i] = 0.0f;

  // Workgroup-relative LDS byte offsets (flat-aperture low 32 bits).
  unsigned stage_off[3];
  stage_off[0] = (unsigned)(uintptr_t)(&stage[0][0]);
  stage_off[1] = (unsigned)(uintptr_t)(&stage[1][0]);
  stage_off[2] = (unsigned)(uintptr_t)(&stage[2][0]);

  // One tile per step = 512 n-rows x 16 batch entries (32 B each): 1024 16-byte
  // chunks; 4 async b128 loads per thread, per-lane addresses.
  // global byte offset of chunk q: n*BB*2 + bbase*2 + half*16, n=q>>1, half=q&1.
  // LDS byte offset of chunk q: q*16.
  auto issue_tile = [&](int buf, int t) {
    unsigned long long gbase =
        (unsigned long long)(uintptr_t)(pre + (size_t)t * G4 * BB);
    #pragma unroll
    for (int j = 0; j < 4; ++j) {
      int q = tid + j * 256;
      unsigned o    = (unsigned)((q >> 1) * (BB * 2) + (q & 1) * 16 + bbase * 2);
      unsigned ldsa = stage_off[buf] + (unsigned)(q * 16);
      asm volatile("global_load_async_to_lds_b128 %0, %1, %2"
                   :: "v"(ldsa), "v"(o), "s"(gbase)
                   : "memory");
    }
  };

  int t0 = dir ? (TT - 1) : 0;
  issue_tile(0, t0);
  int cur = 0;

  for (int step = 0; step < TT; ++step) {
    int t    = dir ? (TT - 1 - step) : step;
    int hcur = step & 1;
    int nx   = cur + 1; if (nx == 3) nx = 0;

    if (step + 1 < TT) {
      issue_tile(nx, dir ? (t - 1) : (t + 1));
      // 4 just-issued loads may stay in flight; previous 4 (current tile) complete.
      asm volatile("s_wait_asynccnt 0x4" ::: "memory");
    } else {
      asm volatile("s_wait_asynccnt 0x0" ::: "memory");
    }
    __syncthreads();   // stage[cur] visible; prev-step h writes visible

    // accumulators from staged pre tile: one ds b128 per gate (8 rows contiguous)
    const _Float16* ps = &stage[cur][0];
    v8f acc[4];
    #pragma unroll
    for (int g = 0; g < 4; ++g) {
      v8h pv = *(const v8h*)(ps + (size_t)(g * HDIM + hc) * 16 + mh);
      #pragma unroll
      for (int i = 0; i < 8; ++i) acc[g][i] = (float)pv[i];
    }

    #pragma unroll
    for (int kt = 0; kt < 4; ++kt) {
      v16h a = load_frag_f16(&h_lds[hcur][0] + kt * 32, HDIM, lane);
      #pragma unroll
      for (int g = 0; g < 4; ++g)
        acc[g] = __builtin_amdgcn_wmma_f32_16x16x32_f16(false, a, false, Bf[g][kt],
                                                        (short)0, acc[g], false, false);
    }

    v8f hv;
    #pragma unroll
    for (int i = 0; i < 8; ++i) {
      float ig = sigf(acc[0][i]);
      float fg = sigf(acc[1][i]);
      float gg = tanh_fast(acc[2][i]);
      float og = sigf(acc[3][i]);
      c[i]  = fg * c[i] + ig * gg;
      hv[i] = og * tanh_fast(c[i]);
    }

    // write h into the other buffer (readers of h_lds[hcur] are pre-barrier-next)
    #pragma unroll
    for (int i = 0; i < 8; ++i) {
      _Float16 h16 = (_Float16)hv[i];
      int m = mh + i;
      h_lds[hcur ^ 1][m * HDIM + hc] = h16;
      out[((size_t)(bbase + m) * TT + t) * HH + coloff + hc] = h16;
    }
    cur = nx;
  }
}

// ---------------------------------------------------------------- LayerNorm + attention score
__global__ __launch_bounds__(256) void ln_attn_kernel(
    const _Float16* __restrict__ y, const float* __restrict__ ln_g,
    const float* __restrict__ ln_b, const float* __restrict__ attn_w,
    const float* __restrict__ attn_b, const unsigned char* __restrict__ mask,
    float* __restrict__ lnout, float* __restrict__ scores)
{
  __shared__ float red[256];
  int row = blockIdx.x;                     // b*T + t
  int ch  = threadIdx.x;                    // 0..255
  float v = (float)y[(size_t)row * HH + ch];

  red[ch] = v; __syncthreads();
  for (int s = 128; s > 0; s >>= 1) { if (ch < s) red[ch] += red[ch + s]; __syncthreads(); }
  float mu = red[0] * (1.0f / HH); __syncthreads();

  float d = v - mu;
  red[ch] = d * d; __syncthreads();
  for (int s = 128; s > 0; s >>= 1) { if (ch < s) red[ch] += red[ch + s]; __syncthreads(); }
  float var = red[0] * (1.0f / HH); __syncthreads();

  float nrm = d * rsqrtf(var + 1e-5f) * ln_g[ch] + ln_b[ch];
  lnout[(size_t)row * HH + ch] = nrm;

  red[ch] = nrm * attn_w[ch]; __syncthreads();
  for (int s = 128; s > 0; s >>= 1) { if (ch < s) red[ch] += red[ch + s]; __syncthreads(); }
  if (ch == 0) {
    int t = row & (TT - 1);
    float s = red[0] + attn_b[0];
    bool mk = (t == 0) || (mask[row] != 0);  // mask[:,0] forced True
    scores[row] = mk ? s : -1e9f;
  }
}

// ---------------------------------------------------------------- softmax over T per batch
__global__ __launch_bounds__(256) void softmax_kernel(const float* __restrict__ scores,
                                                      float* __restrict__ wts) {
  __shared__ float red[256];
  int b = blockIdx.x, tid = threadIdx.x;
  float m = -1e30f;
  for (int t = tid; t < TT; t += 256) m = fmaxf(m, scores[b * TT + t]);
  red[tid] = m; __syncthreads();
  for (int s = 128; s > 0; s >>= 1) { if (tid < s) red[tid] = fmaxf(red[tid], red[tid + s]); __syncthreads(); }
  m = red[0]; __syncthreads();
  float sum = 0.0f;
  for (int t = tid; t < TT; t += 256) sum += fexp(scores[b * TT + t] - m);
  red[tid] = sum; __syncthreads();
  for (int s = 128; s > 0; s >>= 1) { if (tid < s) red[tid] += red[tid + s]; __syncthreads(); }
  float Zi = frcp(red[0]);
  for (int t = tid; t < TT; t += 256) wts[b * TT + t] = fexp(scores[b * TT + t] - m) * Zi;
}

// ---------------------------------------------------------------- weighted pool
__global__ void pooled_kernel(const float* __restrict__ ln, const float* __restrict__ wts,
                              float* __restrict__ pooled) {
  int b = blockIdx.x, ch = threadIdx.x;     // 256 threads
  float acc = 0.0f;
  for (int t = 0; t < TT; ++t)
    acc += wts[b * TT + t] * ln[((size_t)b * TT + t) * HH + ch];
  pooled[b * HH + ch] = acc;
}

// ---------------------------------------------------------------- h2 = ln + pooled (f16)
__global__ void addpool_kernel(const float* __restrict__ ln, const float* __restrict__ pooled,
                               _Float16* __restrict__ h2) {
  size_t i = (size_t)blockIdx.x * 256 + threadIdx.x;   // over NROW*HH
  int ch = (int)(i & (HH - 1));
  int b  = (int)(i >> 18);                              // T*H = 2^18
  h2[i] = (_Float16)(ln[i] + pooled[b * HH + ch]);
}

// ---------------------------------------------------------------- fc2 -> emissions (N=20)
__global__ __launch_bounds__(256) void fc2_kernel(const _Float16* __restrict__ a1,
                                                  const float* __restrict__ w,
                                                  const float* __restrict__ bias,
                                                  float* __restrict__ em) {
  int lane = threadIdx.x & 31;
  int row  = blockIdx.x * 8 + (threadIdx.x >> 5);
  if (lane >= KK) return;
  const _Float16* xr = a1 + (size_t)row * HH;
  float acc = bias[lane];
  for (int j = 0; j < HH; ++j) acc += (float)xr[j] * w[lane * HH + j];
  em[(size_t)row * KK + lane] = acc;
}

// ---------------------------------------------------------------- CRF numerator
__global__ void crf_num_kernel(const int* __restrict__ tags, const unsigned char* __restrict__ mask,
                               const float* __restrict__ em, const float* __restrict__ start,
                               const float* __restrict__ endv, const float* __restrict__ trans,
                               float* __restrict__ num) {
  int b = threadIdx.x;
  if (b >= BB) return;
  const int* tg = tags + b * TT;
  const unsigned char* mk = mask + b * TT;
  int t0 = tg[0];
  float s = start[t0] + em[((size_t)b * TT) * KK + t0];
  int prev = t0, last = 0;
  for (int t = 1; t < TT; ++t) {
    int tgt = tg[t];
    if (mk[t]) { s += trans[prev * KK + tgt] + em[((size_t)b * TT + t) * KK + tgt]; last = t; }
    prev = tgt;
  }
  s += endv[tg[last]];
  num[b] = s;
}

// ---------------------------------------------------------------- CRF forward (logZ), wave32 shuffles
__global__ __launch_bounds__(32) void crf_fwd_kernel(const unsigned char* __restrict__ mask,
                                                     const float* __restrict__ em,
                                                     const float* __restrict__ start,
                                                     const float* __restrict__ endv,
                                                     const float* __restrict__ trans,
                                                     float* __restrict__ logZ) {
  __shared__ float tr[KK * KK];
  int b = blockIdx.x, lane = threadIdx.x;
  for (int i = lane; i < KK * KK; i += 32) tr[i] = trans[i];
  __syncthreads();
  int j = (lane < KK) ? lane : (KK - 1);
  float alpha = start[j] + em[((size_t)b * TT) * KK + j];
  if (lane >= KK) alpha = -1e30f;
  for (int t = 1; t < TT; ++t) {
    float e = em[((size_t)b * TT + t) * KK + j];
    float m = -1e30f;
    #pragma unroll
    for (int i = 0; i < KK; ++i) {
      float ai = __shfl(alpha, i, 32);
      m = fmaxf(m, ai + tr[i * KK + j]);
    }
    float ssum = 0.0f;
    #pragma unroll
    for (int i = 0; i < KK; ++i) {
      float ai = __shfl(alpha, i, 32);
      ssum += fexp(ai + tr[i * KK + j] - m);
    }
    float nxt = m + flog(ssum) + e;
    bool mk = mask[b * TT + t] != 0;
    if (lane < KK) alpha = mk ? nxt : alpha;
  }
  float v = (lane < KK) ? (alpha + endv[j]) : -1e30f;
  float m = v;
  for (int off = 16; off > 0; off >>= 1) m = fmaxf(m, __shfl_xor(m, off, 32));
  float s = (lane < KK) ? fexp(v - m) : 0.0f;
  for (int off = 16; off > 0; off >>= 1) s += __shfl_xor(s, off, 32);
  if (lane == 0) logZ[b] = m + flog(s);
}

// ---------------------------------------------------------------- final NLL reduce
__global__ void final_kernel(const float* __restrict__ num, const float* __restrict__ logZ,
                             float* __restrict__ out) {
  __shared__ float red[64];
  int b = threadIdx.x;
  red[b] = num[b] - logZ[b];
  __syncthreads();
  for (int s = 32; s > 0; s >>= 1) { if (b < s) red[b] += red[b + s]; __syncthreads(); }
  if (b == 0) out[0] = -red[0];
}

// ================================================================ host launcher
extern "C" void kernel_launch(void* const* d_in, const int* in_sizes, int n_in,
                              void* d_out, int out_size, void* d_ws, size_t ws_size,
                              hipStream_t stream) {
  (void)in_sizes; (void)n_in; (void)out_size; (void)ws_size;
  const int*           x    = (const int*)d_in[0];
  const int*           tags = (const int*)d_in[1];
  const unsigned char* mask = (const unsigned char*)d_in[2];   // numpy bool = 1 byte
  const float*         emb  = (const float*)d_in[3];
  const float* w_ih[4] = { (const float*)d_in[4],  (const float*)d_in[8],
                           (const float*)d_in[12], (const float*)d_in[16] };
  const float* w_hh[4] = { (const float*)d_in[5],  (const float*)d_in[9],
                           (const float*)d_in[13], (const float*)d_in[17] };
  const float* b_ih[4] = { (const float*)d_in[6],  (const float*)d_in[10],
                           (const float*)d_in[14], (const float*)d_in[18] };
  const float* b_hh[4] = { (const float*)d_in[7],  (const float*)d_in[11],
                           (const float*)d_in[15], (const float*)d_in[19] };
  const float* ln_g   = (const float*)d_in[20];
  const float* ln_b   = (const float*)d_in[21];
  const float* attn_w = (const float*)d_in[22];
  const float* attn_b = (const float*)d_in[23];
  const float* fc1_w  = (const float*)d_in[24];
  const float* fc1_b  = (const float*)d_in[25];
  const float* fc2_w  = (const float*)d_in[26];
  const float* fc2_b  = (const float*)d_in[27];
  const float* crf_start = (const float*)d_in[28];
  const float* crf_end   = (const float*)d_in[29];
  const float* crf_trans = (const float*)d_in[30];

  // ---- workspace carving (~225 MB total; pre buffers re-used post-LSTM) ----
  char* ws = (char*)d_ws;
  size_t off = 0;
  auto carve = [&](size_t bytes) -> char* {
    char* p = ws + off;
    off += (bytes + 255) & ~(size_t)255;
    return p;
  };
  _Float16* emb16  = (_Float16*)carve((size_t)NROW * EE * 2);   // 16 MB
  _Float16* w16_0  = (_Float16*)carve((size_t)G4 * EE * 2);
  _Float16* w16_1  = (_Float16*)carve((size_t)G4 * EE * 2);
  _Float16* w16_2  = (_Float16*)carve((size_t)G4 * HH * 2);
  _Float16* w16_3  = (_Float16*)carve((size_t)G4 * HH * 2);
  _Float16* fc1w16 = (_Float16*)carve((size_t)HH * HH * 2);
  _Float16* x1     = (_Float16*)carve((size_t)NROW * HH * 2);   // 32 MB layer-1 input
  _Float16* yv     = (_Float16*)carve((size_t)NROW * HH * 2);   // 32 MB layer-1 output
  char*     preA   = carve((size_t)TT * BB * G4 * 2);           // 64 MB
  char*     preB   = carve((size_t)TT * BB * G4 * 2);           // 64 MB
  float* scores = (float*)carve((size_t)NROW * 4);
  float* wts    = (float*)carve((size_t)NROW * 4);
  float* pooled = (float*)carve((size_t)BB * HH * 4);
  float* emis   = (float*)carve((size_t)NROW * KK * 4);
  float* numv   = (float*)carve((size_t)BB * 4);
  float* logZv  = (float*)carve((size_t)BB * 4);
  // post-LSTM aliases of pre buffers
  _Float16* pre_f = (_Float16*)preA;
  _Float16* pre_r = (_Float16*)preB;
  float*    lnout = (float*)preA;                                // 64 MB
  _Float16* h2    = (_Float16*)preB;                             // 32 MB
  _Float16* a1    = (_Float16*)(preB + (size_t)NROW * HH * 2);   // 32 MB

  // ---- weight casts ----
  cast16_kernel<<<(G4 * EE + 255) / 256, 256, 0, stream>>>(w_ih[0], w16_0, G4 * EE);
  cast16_kernel<<<(G4 * EE + 255) / 256, 256, 0, stream>>>(w_ih[1], w16_1, G4 * EE);
  cast16_kernel<<<(G4 * HH + 255) / 256, 256, 0, stream>>>(w_ih[2], w16_2, G4 * HH);
  cast16_kernel<<<(G4 * HH + 255) / 256, 256, 0, stream>>>(w_ih[3], w16_3, G4 * HH);
  cast16_kernel<<<(HH * HH + 255) / 256, 256, 0, stream>>>(fc1_w, fc1w16, HH * HH);

  // ---- embedding ----
  embed_kernel<<<NROW, EE, 0, stream>>>(x, emb, emb16);

  // ---- layer 0: input projections + recurrence ----
  wmma_gemm_kernel<4, 1, 0><<<NROW / 16, 256, 0, stream>>>(emb16, w16_0, b_ih[0], b_hh[0], pre_f, EE, G4);
  wmma_gemm_kernel<4, 1, 0><<<NROW / 16, 256, 0, stream>>>(emb16, w16_1, b_ih[1], b_hh[1], pre_r, EE, G4);
  lstm_rec_kernel<<<8, 256, 0, stream>>>(pre_f, pre_r, w_hh[0], w_hh[1], x1);

  // ---- layer 1 ----
  wmma_gemm_kernel<4, 1, 0><<<NROW / 16, 256, 0, stream>>>(x1, w16_2, b_ih[2], b_hh[2], pre_f, HH, G4);
  wmma_gemm_kernel<4, 1, 0><<<NROW / 16, 256, 0, stream>>>(x1, w16_3, b_ih[3], b_hh[3], pre_r, HH, G4);
  lstm_rec_kernel<<<8, 256, 0, stream>>>(pre_f, pre_r, w_hh[2], w_hh[3], yv);

  // ---- LayerNorm + attention pooling ----
  ln_attn_kernel<<<NROW, 256, 0, stream>>>(yv, ln_g, ln_b, attn_w, attn_b, mask, lnout, scores);
  softmax_kernel<<<BB, 256, 0, stream>>>(scores, wts);
  pooled_kernel<<<BB, HH, 0, stream>>>(lnout, wts, pooled);
  addpool_kernel<<<(NROW * HH) / 256, 256, 0, stream>>>(lnout, pooled, h2);

  // ---- fc1 (WMMA + ReLU) and fc2 (emissions) ----
  wmma_gemm_kernel<2, 0, 1><<<NROW / 16, 256, 0, stream>>>(h2, fc1w16, fc1_b, nullptr, a1, HH, HH);
  fc2_kernel<<<NROW / 8, 256, 0, stream>>>(a1, fc2_w, fc2_b, emis);

  // ---- CRF ----
  crf_num_kernel<<<1, 64, 0, stream>>>(tags, mask, emis, crf_start, crf_end, crf_trans, numv);
  crf_fwd_kernel<<<BB, 32, 0, stream>>>(mask, emis, crf_start, crf_end, crf_trans, logZv);
  final_kernel<<<1, 64, 0, stream>>>(numv, logZv, (float*)d_out);
}
